// MultiHeadAttention_89575837926001
// MI455X (gfx1250) — matrix operand order repaired and checked by reading, not compiled, not run on hardware
//
#include <hip/hip_runtime.h>

typedef __attribute__((ext_vector_type(16))) _Float16 v16h;
typedef __attribute__((ext_vector_type(8)))  float    v8f;
typedef __attribute__((ext_vector_type(8)))  _Float16 h8;   // 16B
typedef __attribute__((ext_vector_type(4)))  _Float16 h4;   // 8B
typedef __attribute__((ext_vector_type(4)))  int      v4i;  // 16B

typedef __attribute__((address_space(1))) v4i gv4i;  // global <4 x i32>
typedef __attribute__((address_space(3))) v4i lv4i;  // LDS    <4 x i32>

#if defined(__has_builtin)
#if __has_builtin(__builtin_amdgcn_global_load_async_to_lds_b128)
#define HAVE_ASYNC_LDS 1
#endif
#endif
#ifndef HAVE_ASYNC_LDS
#define HAVE_ASYNC_LDS 0
#endif

// 16-byte global -> LDS copy: async (no VGPR round-trip, ASYNCcnt-tracked) when
// the toolchain exposes the gfx1250 builtin, else a b128 register copy.
__device__ __forceinline__ void copy16_g2l(_Float16* l, const _Float16* g) {
#if HAVE_ASYNC_LDS
  __builtin_amdgcn_global_load_async_to_lds_b128((gv4i*)g, (lv4i*)l, 0, 0);
#else
  *(h8*)l = *(const h8*)g;
#endif
}

// Wait for this wave's async tile transfers, then workgroup barrier.
__device__ __forceinline__ void tile_sync() {
#if HAVE_ASYNC_LDS
#if __has_builtin(__builtin_amdgcn_s_wait_asynccnt)
  __builtin_amdgcn_s_wait_asynccnt(0);
#else
  asm volatile("s_wait_asynccnt 0" ::: "memory");
#endif
#endif
  __syncthreads();
}

// ---------------------------------------------------------------------------
// WMMA fragment gathers from LDS, contiguous k = 16*(lane/16) + e per lane (wave32).
// ---------------------------------------------------------------------------

// A-matrix 16x32 (MxK), 16-bit.
__device__ __forceinline__ v16h frag_a(const _Float16* base, int ld, int lane) {
  v16h a;
  const int m  = lane & 15;
  const int kb = (lane < 16) ? 0 : 16;
#pragma unroll
  for (int e = 0; e < 16; ++e) {
    const int k = kb + e;
    a[e] = base[m * ld + k];
  }
  return a;
}

// B-matrix 32x16 (KxN), K-major tile in LDS.
__device__ __forceinline__ v16h frag_b(const _Float16* base, int ld, int lane) {
  v16h b;
  const int n  = lane & 15;
  const int kb = (lane < 16) ? 0 : 16;
#pragma unroll
  for (int e = 0; e < 16; ++e) {
    b[e] = base[(kb + e) * ld + n];
  }
  return b;
}

// B fragment from a row-major tile read transposed (for Q.K^T).
__device__ __forceinline__ v16h frag_bt(const _Float16* base, int ld, int lane) {
  v16h b;
  const int n  = lane & 15;
  const int kb = (lane < 16) ? 0 : 16;
#pragma unroll
  for (int e = 0; e < 16; ++e) {
    b[e] = base[n * ld + kb + e];
  }
  return b;
}

// ---------------------------------------------------------------------------
// f32 -> f16 conversion (weights pre-pass), 4 elems/thread, vectorized.
// ---------------------------------------------------------------------------
__global__ __launch_bounds__(256) void cvt_f16_kernel(const float* __restrict__ in,
                                                      _Float16* __restrict__ out, int n) {
  const int i = (blockIdx.x * 256 + threadIdx.x) * 4;
  if (i + 3 < n) {
    const float4 v = *(const float4*)(in + i);
    h4 o = {(_Float16)v.x, (_Float16)v.y, (_Float16)v.z, (_Float16)v.w};
    *(h4*)(out + i) = o;
  }
}

// ---------------------------------------------------------------------------
// GEMM: C[MxN] = f16(X) @ Wh + bias. 64x64 tile, 128 thr (4 waves).
// InT = float (convert on stage-in) or _Float16 (async b128 stage-in).
// HEADS=true : f16 out in [B][H][S][64] (split heads); else f32 row-major.
// ---------------------------------------------------------------------------
template <bool HEADS, typename InT>
__global__ __launch_bounds__(128) void wmma_gemm_kernel(
    const InT* __restrict__ X, const _Float16* __restrict__ Wh,
    const float* __restrict__ bias, void* __restrict__ OutP,
    int M, int N, int K, int S) {
  constexpr int LDA = 40;  // 32 + 8 pad (halves): 8B-aligned rows, 16B chunks
  constexpr int LDB = 72;  // 64 + 8 pad
  __shared__ _Float16 sA[64 * LDA];
  __shared__ _Float16 sB[32 * LDB];

  const int tid  = threadIdx.x;
  const int lane = tid & 31;
  const int wid  = tid >> 5;
  const int n0   = blockIdx.x * 64;
  const int m0   = blockIdx.y * 64;

  const v8f vzero = {0.f, 0.f, 0.f, 0.f, 0.f, 0.f, 0.f, 0.f};
  v8f acc[4];
#pragma unroll
  for (int t = 0; t < 4; ++t) acc[t] = vzero;

  for (int kt = 0; kt < K; kt += 32) {
    // ---- stage A tile 64x32 ----
    if constexpr (sizeof(InT) == 4) {
      for (int idx = tid; idx < 64 * 8; idx += 128) {  // 8 float4 per row
        const int r = idx >> 3, c4 = idx & 7;
        const float4 xv = *(const float4*)(X + (size_t)(m0 + r) * K + kt + c4 * 4);
        h4 hv = {(_Float16)xv.x, (_Float16)xv.y, (_Float16)xv.z, (_Float16)xv.w};
        *(h4*)(sA + r * LDA + c4 * 4) = hv;
      }
    } else {
      for (int idx = tid; idx < 64 * 4; idx += 128) {  // 4 b128 per row
        const int r = idx >> 2, c8 = idx & 3;
        copy16_g2l(sA + r * LDA + c8 * 8, (const _Float16*)X + (size_t)(m0 + r) * K + kt + c8 * 8);
      }
    }
    // ---- stage B tile 32x64 (f16 weights, async path) ----
    for (int idx = tid; idx < 32 * 8; idx += 128) {  // 8 b128 per row
      const int r = idx >> 3, c8 = idx & 7;
      copy16_g2l(sB + r * LDB + c8 * 8, Wh + (size_t)(kt + r) * N + n0 + c8 * 8);
    }
    // prefetch next K-step while we compute
    if (kt + 32 < K) {
      __builtin_prefetch(Wh + (size_t)(kt + 32 + (tid >> 2)) * N + n0, 0, 1);
      __builtin_prefetch(X + (size_t)(m0 + (tid >> 1)) * K + kt + 32, 0, 1);
    }
    tile_sync();

    const v16h a = frag_a(sA + (16 * wid) * LDA, LDA, lane);
#pragma unroll
    for (int t = 0; t < 4; ++t) {
      const v16h b = frag_b(sB + t * 16, LDB, lane);
      acc[t] = __builtin_amdgcn_wmma_f32_16x16x32_f16(
          false, a, false, b, (short)0, acc[t], false, false);
    }
    __syncthreads();
  }

  const int half8 = (lane < 16) ? 0 : 8;
  const int nl    = lane & 15;
#pragma unroll
  for (int t = 0; t < 4; ++t) {
#pragma unroll
    for (int j = 0; j < 8; ++j) {
      const int gm  = m0 + 16 * wid + j + half8;
      const int gn  = n0 + t * 16 + nl;
      const float v = acc[t][j] + bias[gn];
      if constexpr (HEADS) {
        _Float16* Out = (_Float16*)OutP;
        const int bb = gm / S, s = gm - bb * S;
        const int h = gn >> 6, d = gn & 63;
        Out[(((size_t)(bb * 16 + h)) * S + s) * 64 + d] = (_Float16)v;
      } else {
        float* Out = (float*)OutP;
        Out[(size_t)gm * N + gn] = v;
      }
    }
  }
  (void)S;
}

// ---------------------------------------------------------------------------
// Flash attention per (64-row query block, batch*head). 4 waves, 16 rows each.
// Online softmax over 64-key blocks; DEPTH = 64. Output f16 [B][S][H*64],
// scaled by (1/rowsum)*(1/sqrt(dk)) [reference applies inv_sqrt_dk twice].
// ---------------------------------------------------------------------------
__global__ __launch_bounds__(128) void flash_attn_kernel(
    const _Float16* __restrict__ Qh, const _Float16* __restrict__ Kh,
    const _Float16* __restrict__ Vh, _Float16* __restrict__ Outb, int S) {
  constexpr int LD = 72;  // 64 + 8 pad
  __shared__ _Float16 sQ[64 * LD];
  __shared__ _Float16 sK[64 * LD];
  __shared__ _Float16 sV[64 * LD];
  __shared__ _Float16 sP[4 * 16 * LD];  // per-wave 16x64 P tile

  const int tid  = threadIdx.x;
  const int lane = tid & 31;
  const int wid  = tid >> 5;
  const int qb   = blockIdx.x;
  const int bh   = blockIdx.y;
  const float scale = 0.125f;  // 1/sqrt(64)

  const _Float16* Qg = Qh + ((size_t)bh * S + (size_t)qb * 64) * 64;
  const _Float16* Kg = Kh + (size_t)bh * S * 64;
  const _Float16* Vg = Vh + (size_t)bh * S * 64;

  for (int idx = tid; idx < 64 * 8; idx += 128) {  // 64 rows x 8 b128 chunks
    const int r = idx >> 3, c8 = idx & 7;
    copy16_g2l(sQ + r * LD + c8 * 8, Qg + (size_t)r * 64 + c8 * 8);
  }
  tile_sync();

  const v16h aQ0 = frag_a(sQ + (16 * wid) * LD, LD, lane);
  const v16h aQ1 = frag_a(sQ + (16 * wid) * LD + 32, LD, lane);

  const v8f vzero = {0.f, 0.f, 0.f, 0.f, 0.f, 0.f, 0.f, 0.f};
  v8f accO[4];
#pragma unroll
  for (int t = 0; t < 4; ++t) accO[t] = vzero;
  float rmax[8], rsum[8];
#pragma unroll
  for (int j = 0; j < 8; ++j) { rmax[j] = -3.0e38f; rsum[j] = 0.f; }

  _Float16* sPw = sP + wid * 16 * LD;

  for (int jb = 0; jb < S; jb += 64) {
    for (int idx = tid; idx < 64 * 8; idx += 128) {
      const int r = idx >> 3, c8 = idx & 7;
      copy16_g2l(sK + r * LD + c8 * 8, Kg + (size_t)(jb + r) * 64 + c8 * 8);
      copy16_g2l(sV + r * LD + c8 * 8, Vg + (size_t)(jb + r) * 64 + c8 * 8);
    }
    // prefetch next key/value block into cache while computing this one
    if (jb + 64 < S) {
      const _Float16* pb = (tid & 1) ? Vg : Kg;
      __builtin_prefetch(pb + (size_t)(jb + 64 + (tid >> 1)) * 64, 0, 1);
    }
    tile_sync();

    // scores: 16x64 = Q(16x64) . K^T, chained over D in two 32-chunks
    v8f cS[4];
#pragma unroll
    for (int t = 0; t < 4; ++t) {
      const v16h b0 = frag_bt(sK + (t * 16) * LD, LD, lane);
      const v16h b1 = frag_bt(sK + (t * 16) * LD + 32, LD, lane);
      v8f cacc = vzero;
      cacc = __builtin_amdgcn_wmma_f32_16x16x32_f16(false, aQ0, false, b0,
                                                    (short)0, cacc, false, false);
      cacc = __builtin_amdgcn_wmma_f32_16x16x32_f16(false, aQ1, false, b1,
                                                    (short)0, cacc, false, false);
      cS[t] = cacc;
    }

    // online softmax; row stats reduced across 16-lane half groups (wave32)
#pragma unroll
    for (int j = 0; j < 8; ++j) {
      float pm = -3.0e38f;
#pragma unroll
      for (int t = 0; t < 4; ++t) pm = fmaxf(pm, cS[t][j] * scale);
      pm = fmaxf(pm, __shfl_xor(pm, 1, 32));
      pm = fmaxf(pm, __shfl_xor(pm, 2, 32));
      pm = fmaxf(pm, __shfl_xor(pm, 4, 32));
      pm = fmaxf(pm, __shfl_xor(pm, 8, 32));
      const float nm   = fmaxf(rmax[j], pm);
      const float corr = __expf(rmax[j] - nm);
      rmax[j] = nm;
      float ps = 0.f;
      const int m = j + ((lane < 16) ? 0 : 8);
#pragma unroll
      for (int t = 0; t < 4; ++t) {
        const float p = __expf(cS[t][j] * scale - nm);
        ps += p;
        sPw[m * LD + t * 16 + (lane & 15)] = (_Float16)p;
        accO[t][j] *= corr;
      }
      ps += __shfl_xor(ps, 1, 32);
      ps += __shfl_xor(ps, 2, 32);
      ps += __shfl_xor(ps, 4, 32);
      ps += __shfl_xor(ps, 8, 32);
      rsum[j] = rsum[j] * corr + ps;
    }
    // sP is per-wave: same-wave DS ordering covers the RAW, no barrier needed.

    // O += P(16x64) . V(64x64), chained over key dim in two 32-chunks
#pragma unroll
    for (int kc = 0; kc < 2; ++kc) {
      const v16h aP = frag_a(sPw + 32 * kc, LD, lane);
#pragma unroll
      for (int t = 0; t < 4; ++t) {
        const v16h bV = frag_b(sV + (32 * kc) * LD + t * 16, LD, lane);
        accO[t] = __builtin_amdgcn_wmma_f32_16x16x32_f16(
            false, aP, false, bV, (short)0, accO[t], false, false);
      }
    }
    __syncthreads();  // protect sK/sV before next block's stage-in
  }

  // finalize: out = O * (1/rowsum) * inv_sqrt_dk  -> f16 [B][S][H*64]
  const int nl    = lane & 15;
  const int half8 = (lane < 16) ? 0 : 8;
  const int bb = bh >> 4, h = bh & 15;
#pragma unroll
  for (int j = 0; j < 8; ++j) {
    const float inv = scale / rsum[j];
    const int s = qb * 64 + 16 * wid + j + half8;
    const size_t row = (size_t)bb * S + s;
#pragma unroll
    for (int t = 0; t < 4; ++t) {
      Outb[row * 1024 + h * 64 + t * 16 + nl] = (_Float16)(accO[t][j] * inv);
    }
  }
}

// ---------------------------------------------------------------------------
// Host launcher
// ---------------------------------------------------------------------------
extern "C" void kernel_launch(void* const* d_in, const int* in_sizes, int n_in,
                              void* d_out, int out_size, void* d_ws, size_t ws_size,
                              hipStream_t stream) {
  const float* q  = (const float*)d_in[0];
  const float* k  = (const float*)d_in[1];
  const float* v  = (const float*)d_in[2];
  const float* wq = (const float*)d_in[3];
  const float* bq = (const float*)d_in[4];
  const float* wk = (const float*)d_in[5];
  const float* bk = (const float*)d_in[6];
  const float* wv = (const float*)d_in[7];
  const float* bv = (const float*)d_in[8];
  const float* wo = (const float*)d_in[9];
  const float* bo = (const float*)d_in[10];

  const int B = 2, S = 2048, DM = 1024, H = 16;
  const int M = B * S;  // 4096

  // workspace layout (f16 unless noted):
  //   Qh/Kh/Vh [B*H][S][64]     3 x 8 MB
  //   wqh/wkh/wvh/woh [DM][DM]  4 x 2 MB
  //   attnbuf [B][S][H*64]      8 MB
  _Float16* Qh  = (_Float16*)d_ws;
  _Float16* Kh  = Qh + (size_t)B * H * S * 64;
  _Float16* Vh  = Kh + (size_t)B * H * S * 64;
  _Float16* wqh = Vh + (size_t)B * H * S * 64;
  _Float16* wkh = wqh + (size_t)DM * DM;
  _Float16* wvh = wkh + (size_t)DM * DM;
  _Float16* woh = wvh + (size_t)DM * DM;
  _Float16* attnbuf = woh + (size_t)DM * DM;

  // pre-convert weights to f16 once (each W tile is otherwise re-converted by
  // all 64 M-blocks); also enables async b128 LDS staging in the GEMMs.
  {
    const int n = DM * DM;
    const dim3 cblk(256), cgrid(n / (256 * 4));
    cvt_f16_kernel<<<cgrid, cblk, 0, stream>>>(wq, wqh, n);
    cvt_f16_kernel<<<cgrid, cblk, 0, stream>>>(wk, wkh, n);
    cvt_f16_kernel<<<cgrid, cblk, 0, stream>>>(wv, wvh, n);
    cvt_f16_kernel<<<cgrid, cblk, 0, stream>>>(wo, woh, n);
  }

  const dim3 blk(128);
  const dim3 gproj(DM / 64, M / 64);
  wmma_gemm_kernel<true, float><<<gproj, blk, 0, stream>>>(q, wqh, bq, (void*)Qh, M, DM, DM, S);
  wmma_gemm_kernel<true, float><<<gproj, blk, 0, stream>>>(k, wkh, bk, (void*)Kh, M, DM, DM, S);
  wmma_gemm_kernel<true, float><<<gproj, blk, 0, stream>>>(v, wvh, bv, (void*)Vh, M, DM, DM, S);

  const dim3 gattn(S / 64, B * H);
  flash_attn_kernel<<<gattn, blk, 0, stream>>>(Qh, Kh, Vh, attnbuf, S);

  wmma_gemm_kernel<false, _Float16><<<gproj, blk, 0, stream>>>(attnbuf, woh, bo, d_out, M, DM, DM, S);

  (void)in_sizes; (void)n_in; (void)out_size; (void)ws_size;
}
